// FTTransformerPNAFused_83588653515396
// MI455X (gfx1250) — matrix operand order repaired and checked by reading, not compiled, not run on hardware
//
#include <hip/hip_runtime.h>
#include <hip/hip_bf16.h>
#include <math.h>

typedef __bf16 bf16_t;
typedef bf16_t v16bf __attribute__((ext_vector_type(16)));
typedef float  v8f   __attribute__((ext_vector_type(8)));

namespace {
constexpr int  kC = 128, kNH = 8, kFF = 512, kF = 128;
constexpr int  kB = 4096, kS = 33;
constexpr long kN = 100000, kE = 500000;
constexpr int  kFUSED = kC + 2 * kF;   // 384
constexpr int  kDH = kC / kNH;         // 16
constexpr long kBS = (long)kB * kS;    // 135168
}

#define DEV static __device__ __forceinline__

DEV float wave_sum(float v) {
#pragma unroll
  for (int o = 16; o > 0; o >>= 1) v += __shfl_xor(v, o, 32);
  return v;
}
// monotonic float<->uint mapping so uint atomic max/min == float max/min
DEV unsigned encf(float f) {
  unsigned u = __float_as_uint(f);
  return (u & 0x80000000u) ? ~u : (u | 0x80000000u);
}
DEV float decf(unsigned k) {
  return __uint_as_float((k & 0x80000000u) ? (k & 0x7fffffffu) : ~k);
}

// ---------------------------------------------------------------------------
// bf16 WMMA GEMM:  Y[M x Nout] = act(X[M x K] * W^T + bias) * scale
//   One wave computes a 16x64 output strip (4 N-tiles) so the A fragment is
//   loaded once per K-step and reused across 4 v_wmma issues.
//   X row-major (ldx), W row-major [Nout x K] (ldw). Nout % 64 == 0, K % 32 == 0,
//   M % 16 == 0. Fragment layouts used here (wave32; A/B use contiguous k = 16h + s):
//     A 16x32 bf16 : lane<16 row=lane, K {0..15}; lane>=16 K {16..31}
//     B 32x16 bf16 : lane n holds column n, 16 contiguous K values (half per group)
//     C/D f32      : vgpr r -> row m0 + (lane>>4)*8 + r, col n0 + (lane&15)
// ---------------------------------------------------------------------------
union FragBF { v16bf v; uint4 q[2]; };

__global__ __launch_bounds__(256) void wmma_gemm_kernel(
    const bf16_t* __restrict__ X, int ldx,
    const bf16_t* __restrict__ W, int ldw,
    const float* __restrict__ bias, const float* __restrict__ row_scale,
    float* __restrict__ Yf, bf16_t* __restrict__ Yh, int ldy,
    long M, int K, int act, int accum) {
  const int lane = threadIdx.x & 31;
  const int wave = threadIdx.x >> 5;
  const long m0 = ((long)blockIdx.y * 8 + wave) * 16;
  const int n0 = blockIdx.x * 64;
  if (m0 >= M) return;
  const int l = lane & 15;
  const int hi = lane >> 4;

  const bf16_t* arow = X + (size_t)(m0 + l) * ldx + hi * 16;
  const bf16_t* brow = W + (size_t)(n0 + l) * ldw + hi * 16;
  const size_t bstep = (size_t)16 * ldw;  // next N-tile = 16 rows of W

  v8f acc0 = {0.f, 0.f, 0.f, 0.f, 0.f, 0.f, 0.f, 0.f};
  v8f acc1 = acc0, acc2 = acc0, acc3 = acc0;

  for (int k0 = 0; k0 < K; k0 += 32) {
    FragBF a, b0, b1, b2, b3;
    a.q[0]  = *(const uint4*)(arow + k0);
    a.q[1]  = *(const uint4*)(arow + k0 + 8);
    b0.q[0] = *(const uint4*)(brow + k0);
    b0.q[1] = *(const uint4*)(brow + k0 + 8);
    b1.q[0] = *(const uint4*)(brow + bstep + k0);
    b1.q[1] = *(const uint4*)(brow + bstep + k0 + 8);
    b2.q[0] = *(const uint4*)(brow + 2 * bstep + k0);
    b2.q[1] = *(const uint4*)(brow + 2 * bstep + k0 + 8);
    b3.q[0] = *(const uint4*)(brow + 3 * bstep + k0);
    b3.q[1] = *(const uint4*)(brow + 3 * bstep + k0 + 8);
    if (k0 + 32 < K) {  // gfx1250 near-cache prefetch of next K tiles
      __builtin_prefetch((const void*)(arow + k0 + 32), 0, 3);
      __builtin_prefetch((const void*)(brow + k0 + 32), 0, 3);
      __builtin_prefetch((const void*)(brow + bstep + k0 + 32), 0, 3);
      __builtin_prefetch((const void*)(brow + 2 * bstep + k0 + 32), 0, 3);
      __builtin_prefetch((const void*)(brow + 3 * bstep + k0 + 32), 0, 3);
    }
    acc0 = __builtin_amdgcn_wmma_f32_16x16x32_bf16(false, a.v, false, b0.v,
                                                   (short)0, acc0, false, false);
    acc1 = __builtin_amdgcn_wmma_f32_16x16x32_bf16(false, a.v, false, b1.v,
                                                   (short)0, acc1, false, false);
    acc2 = __builtin_amdgcn_wmma_f32_16x16x32_bf16(false, a.v, false, b2.v,
                                                   (short)0, acc2, false, false);
    acc3 = __builtin_amdgcn_wmma_f32_16x16x32_bf16(false, a.v, false, b3.v,
                                                   (short)0, acc3, false, false);
  }

  v8f accs[4] = {acc0, acc1, acc2, acc3};
  float rs[8];
#pragma unroll
  for (int r = 0; r < 8; ++r)
    rs[r] = row_scale ? row_scale[m0 + hi * 8 + r] : 1.f;
#pragma unroll
  for (int t = 0; t < 4; ++t) {
    const int col = n0 + t * 16 + l;
    const float bv = bias ? bias[col] : 0.f;
#pragma unroll
    for (int r = 0; r < 8; ++r) {
      const long row = m0 + hi * 8 + r;
      float v = accs[t][r] + bv;
      if (act == 1) v = fmaxf(v, 0.f);
      else if (act == 2) v = (v > 0.f) ? v : 0.01f * v;
      v *= rs[r];
      const size_t idx = (size_t)row * ldy + col;
      if (Yh) Yh[idx] = (bf16_t)v;
      else if (accum) Yf[idx] += v;
      else Yf[idx] = v;
    }
  }
}

// ---------------------------------------------------------------------------
__global__ __launch_bounds__(256) void f2bf_kernel(const float* __restrict__ x,
                                                   bf16_t* __restrict__ y, long n) {
  long i = (long)blockIdx.x * 256 + threadIdx.x;
  if (i < n) y[i] = (bf16_t)x[i];
}

__global__ __launch_bounds__(256) void fill_u32_kernel(unsigned* __restrict__ p,
                                                       unsigned v, long n) {
  long i = (long)blockIdx.x * 256 + threadIdx.x;
  if (i < n) p[i] = v;
}

// Wcomb[r][c] = sum_k Wpre[r][2F+k] * We[k][c]      (fold e@We.T@W_e.T)
__global__ __launch_bounds__(256) void fold_we_kernel(const float* __restrict__ Wpre,
                                                      const float* __restrict__ We,
                                                      float* __restrict__ Wcomb) {
  int idx = blockIdx.x * 256 + threadIdx.x;
  if (idx >= kF * kF) return;
  int r = idx / kF, c = idx % kF;
  float s = 0.f;
  for (int k = 0; k < kF; ++k) s += Wpre[r * 3 * kF + 2 * kF + k] * We[k * kF + c];
  Wcomb[idx] = s;
}
__global__ void fold_bcomb_kernel(const float* __restrict__ Wpre,
                                  const float* __restrict__ be,
                                  const float* __restrict__ bpre,
                                  float* __restrict__ bcomb) {
  int r = threadIdx.x;
  if (r >= kF) return;
  float s = 0.f;
  for (int k = 0; k < kF; ++k) s += Wpre[r * 3 * kF + 2 * kF + k] * be[k];
  bcomb[r] = s + bpre[r];
}
// Wfold[r][c] = sum_k Wlin[r][k] * Wpost[k][c]   (fold the 13F projection)
__global__ __launch_bounds__(256) void fold_wpost_kernel(const float* __restrict__ Wlin,
                                                         const float* __restrict__ Wpost,
                                                         float* __restrict__ Wfold) {
  int idx = blockIdx.x * 256 + threadIdx.x;
  if (idx >= kF * 13 * kF) return;
  int r = idx / (13 * kF), c = idx % (13 * kF);
  float s = 0.f;
  for (int k = 0; k < kF; ++k) s += Wlin[r * kF + k] * Wpost[k * 13 * kF + c];
  Wfold[idx] = s;
}
__global__ void fold_bfold_kernel(const float* __restrict__ Wlin,
                                  const float* __restrict__ bpost,
                                  const float* __restrict__ blin,
                                  float* __restrict__ bfold) {
  int r = threadIdx.x;
  if (r >= kF) return;
  float s = 0.f;
  for (int k = 0; k < kF; ++k) s += Wlin[r * kF + k] * bpost[k];
  bfold[r] = s + blin[r];
}

// ---------------------------------------------------------------------------
// Attention: one thread per (b, h, q). S=33, dh=16; register softmax.
__global__ __launch_bounds__(256) void attention_kernel(const bf16_t* __restrict__ qkv,
                                                        bf16_t* __restrict__ o) {
  long idx = (long)blockIdx.x * 256 + threadIdx.x;
  if (idx >= (long)kB * kNH * kS) return;
  int q = (int)(idx % kS);
  long t = idx / kS;
  int h = (int)(t % kNH);
  long b = t / kNH;

  const bf16_t* qrow = qkv + (size_t)(b * kS + q) * (3 * kC) + h * kDH;
  float qv[kDH];
#pragma unroll
  for (int d = 0; d < kDH; ++d) qv[d] = (float)qrow[d];

  float sc[kS];
  float mx = -3.4e38f;
  for (int k = 0; k < kS; ++k) {
    const bf16_t* krow = qkv + (size_t)(b * kS + k) * (3 * kC) + kC + h * kDH;
    float s = 0.f;
#pragma unroll
    for (int d = 0; d < kDH; ++d) s += qv[d] * (float)krow[d];
    s *= 0.25f;  // 1/sqrt(16)
    sc[k] = s;
    mx = fmaxf(mx, s);
  }
  float sum = 0.f;
  for (int k = 0; k < kS; ++k) { sc[k] = expf(sc[k] - mx); sum += sc[k]; }
  const float inv = 1.f / sum;

  float out[kDH];
#pragma unroll
  for (int d = 0; d < kDH; ++d) out[d] = 0.f;
  for (int k = 0; k < kS; ++k) {
    const bf16_t* vrow = qkv + (size_t)(b * kS + k) * (3 * kC) + 2 * kC + h * kDH;
    float p = sc[k] * inv;
#pragma unroll
    for (int d = 0; d < kDH; ++d) out[d] += p * (float)vrow[d];
  }
  bf16_t* orow = o + (size_t)(b * kS + q) * kC + h * kDH;
#pragma unroll
  for (int d = 0; d < kDH; ++d) orow[d] = (bf16_t)out[d];
}

// ---------------------------------------------------------------------------
// Generic wave-per-row LayerNorm: Y = ln(X (+R)) * g + b. cols multiple of 32, <=512.
__global__ __launch_bounds__(256) void ln_rows_kernel(
    const float* __restrict__ X, const float* __restrict__ R,
    const float* __restrict__ g, const float* __restrict__ bb,
    float* __restrict__ Yf, bf16_t* __restrict__ Yh, long rows, int cols) {
  long row = (long)blockIdx.x * 8 + (threadIdx.x >> 5);
  if (row >= rows) return;
  const int lane = threadIdx.x & 31;
  const int per = cols >> 5;
  float v[16];
  const float* xr = X + (size_t)row * cols;
  const float* rr = R ? R + (size_t)row * cols : nullptr;
  float s = 0.f;
  for (int j = 0; j < per; ++j) {
    float t = xr[lane * per + j];
    if (rr) t += rr[lane * per + j];
    v[j] = t; s += t;
  }
  s = wave_sum(s) / cols;
  float q = 0.f;
  for (int j = 0; j < per; ++j) { v[j] -= s; q += v[j] * v[j]; }
  q = wave_sum(q) / cols;
  const float rs = rsqrtf(q + 1e-5f);
  for (int j = 0; j < per; ++j) {
    int c = lane * per + j;
    float o = v[j] * rs * g[c] + bb[c];
    if (Yf) Yf[(size_t)row * cols + c] = o;
    if (Yh) Yh[(size_t)row * cols + c] = (bf16_t)o;
  }
}

// Double LN for the transformer tail: Y = ln( ln(X+R, g1,b1), g2,b2 ), cols = 128.
__global__ __launch_bounds__(256) void ln2x_kernel(
    const float* __restrict__ X, const float* __restrict__ R,
    const float* __restrict__ g1, const float* __restrict__ b1,
    const float* __restrict__ g2, const float* __restrict__ b2,
    float* __restrict__ Y, long rows) {
  long row = (long)blockIdx.x * 8 + (threadIdx.x >> 5);
  if (row >= rows) return;
  const int lane = threadIdx.x & 31;
  float v[4];
  const float* xr = X + (size_t)row * kC;
  const float* rr = R + (size_t)row * kC;
  float s = 0.f;
#pragma unroll
  for (int j = 0; j < 4; ++j) { v[j] = xr[lane * 4 + j] + rr[lane * 4 + j]; s += v[j]; }
  s = wave_sum(s) * (1.f / kC);
  float q = 0.f;
#pragma unroll
  for (int j = 0; j < 4; ++j) { v[j] -= s; q += v[j] * v[j]; }
  q = wave_sum(q) * (1.f / kC);
  float rs = rsqrtf(q + 1e-5f);
#pragma unroll
  for (int j = 0; j < 4; ++j) { int c = lane * 4 + j; v[j] = v[j] * rs * g1[c] + b1[c]; }
  s = 0.f;
#pragma unroll
  for (int j = 0; j < 4; ++j) s += v[j];
  s = wave_sum(s) * (1.f / kC);
  q = 0.f;
#pragma unroll
  for (int j = 0; j < 4; ++j) { v[j] -= s; q += v[j] * v[j]; }
  q = wave_sum(q) * (1.f / kC);
  rs = rsqrtf(q + 1e-5f);
#pragma unroll
  for (int j = 0; j < 4; ++j) {
    int c = lane * 4 + j;
    Y[(size_t)row * kC + c] = v[j] * rs * g2[c] + b2[c];
  }
}

// ---------------------------------------------------------------------------
// GNN edge pass: one wave per edge; msg = xiW[dst] + xjW[src] + ewc[e]; atomic aggr.
__global__ __launch_bounds__(256) void edge_accum_kernel(
    const float* __restrict__ xiW, const float* __restrict__ xjW,
    const bf16_t* __restrict__ ewc, const int* __restrict__ src,
    const int* __restrict__ dst, float* __restrict__ s1, float* __restrict__ s2,
    unsigned* __restrict__ mxe, unsigned* __restrict__ mne,
    float* __restrict__ deg) {
  long e = (long)blockIdx.x * 8 + (threadIdx.x >> 5);
  if (e >= kE) return;
  const int lane = threadIdx.x & 31;
  const int d = dst[e], s = src[e];
  const float* xi = xiW + (size_t)d * kF;
  const float* xj = xjW + (size_t)s * kF;
  const bf16_t* ew = ewc + (size_t)e * kF;
#pragma unroll
  for (int i = lane; i < kF; i += 32) {
    const float m = xi[i] + xj[i] + (float)ew[i];
    const size_t o = (size_t)d * kF + i;
    atomicAdd(&s1[o], m);
    atomicAdd(&s2[o], m * m);
    atomicMax(&mxe[o], encf(m));
    atomicMin(&mne[o], encf(m));
  }
  if (lane == 0) atomicAdd(&deg[d], 1.f);
}

__global__ __launch_bounds__(256) void deg_log_sum_kernel(const float* __restrict__ deg,
                                                          float* __restrict__ dsum, long n) {
  long i = (long)blockIdx.x * 256 + threadIdx.x;
  float v = (i < n) ? logf(deg[i] + 1.f) : 0.f;
  v = wave_sum(v);
  if ((threadIdx.x & 31) == 0) atomicAdd(dsum, v);
}

// Build aggr = [mean|max|min|std] (bf16) + amp / 1/amp.
__global__ __launch_bounds__(256) void aggr_kernel(
    const float* __restrict__ s1, const float* __restrict__ s2,
    const unsigned* __restrict__ mxe, const unsigned* __restrict__ mne,
    const float* __restrict__ deg, const float* __restrict__ dsum,
    bf16_t* __restrict__ aggr, float* __restrict__ amp,
    float* __restrict__ inv_amp) {
  long node = (long)blockIdx.x * 8 + (threadIdx.x >> 5);
  if (node >= kN) return;
  const int lane = threadIdx.x & 31;
  const float dg = deg[node];
  const float dc = fmaxf(dg, 1.f);
  const bool has = dg > 0.f;
  const float avg = dsum[0] / (float)kN;
  const float a = logf(dc + 1.f) / avg;
  if (lane == 0) { amp[node] = a; inv_amp[node] = 1.f / a; }
  bf16_t* ar = aggr + (size_t)node * (4 * kF);
  const size_t base = (size_t)node * kF;
#pragma unroll
  for (int i = lane; i < kF; i += 32) {
    const float m1 = s1[base + i] / dc;
    const float m2 = s2[base + i] / dc;
    const float sd = sqrtf(fmaxf(m2 - m1 * m1, 0.f) + 1e-5f);
    const float xm = has ? decf(mxe[base + i]) : 0.f;
    const float xn = has ? decf(mne[base + i]) : 0.f;
    ar[i] = (bf16_t)m1;
    ar[kF + i] = (bf16_t)xm;
    ar[2 * kF + i] = (bf16_t)xn;
    ar[3 * kF + i] = (bf16_t)sd;
  }
}

// BatchNorm column stats (per-block partial + atomic).
__global__ void colstats_kernel(const float* __restrict__ gout,
                                float* __restrict__ colsum,
                                float* __restrict__ colsq, long rows) {
  const int col = threadIdx.x;  // blockDim = 128
  long r0 = (long)blockIdx.x * 128;
  long rend = r0 + 128; if (rend > rows) rend = rows;
  float s = 0.f, q = 0.f;
  for (long r = r0; r < rend; ++r) {
    float v = gout[(size_t)r * kF + col];
    s += v; q += v * v;
  }
  atomicAdd(&colsum[col], s);
  atomicAdd(&colsq[col], q);
}

__global__ __launch_bounds__(256) void bn_final_kernel(
    const float* __restrict__ gout, const float* __restrict__ x_gnn,
    const float* __restrict__ colsum, const float* __restrict__ colsq,
    const float* __restrict__ bn_g, const float* __restrict__ bn_b,
    float* __restrict__ xg_out) {
  long i = (long)blockIdx.x * 256 + threadIdx.x;
  if (i >= kN * kF) return;
  const int col = (int)(i & (kF - 1));
  const float bm = colsum[col] / (float)kN;
  const float bv = colsq[col] / (float)kN - bm * bm;
  const float v = (gout[i] - bm) * rsqrtf(bv + 1e-5f) * bn_g[col] + bn_b[col];
  xg_out[i] = (x_gnn[i] + fmaxf(v, 0.f)) * 0.5f;
}

// ---------------------------------------------------------------------------
// Fusion: xf = [x_cls | xg[src] | xg[dst]]
__global__ __launch_bounds__(256) void fuse_build_kernel(
    const float* __restrict__ x_tabn, const float* __restrict__ xg,
    const int* __restrict__ src, const int* __restrict__ dst,
    float* __restrict__ xf) {
  int b = blockIdx.x * 8 + (threadIdx.x >> 5);
  if (b >= kB) return;
  const int lane = threadIdx.x & 31;
  const float* cls = x_tabn + (size_t)b * kS * kC;  // row (b, 0)
  float* row = xf + (size_t)b * kFUSED;
#pragma unroll
  for (int i = lane; i < kC; i += 32) row[i] = cls[i];
  const int s = src[b], d = dst[b];
#pragma unroll
  for (int i = lane; i < kF; i += 32) {
    row[kC + i] = xg[(size_t)s * kF + i];
    row[kC + kF + i] = xg[(size_t)d * kF + i];
  }
}

// xfo = (xf + ln(h3,fon))/2 ; write tab row (b,0); keep xfo in XF for scatters.
__global__ __launch_bounds__(256) void fuse_final_kernel(
    const float* __restrict__ H3, float* __restrict__ XF,
    const float* __restrict__ g, const float* __restrict__ bb,
    float* __restrict__ out_tab) {
  int b = blockIdx.x * 8 + (threadIdx.x >> 5);
  if (b >= kB) return;
  const int lane = threadIdx.x & 31;
  const float* hr = H3 + (size_t)b * kFUSED;
  float v[12];
  float s = 0.f;
#pragma unroll
  for (int j = 0; j < 12; ++j) { v[j] = hr[lane * 12 + j]; s += v[j]; }
  s = wave_sum(s) * (1.f / kFUSED);
  float q = 0.f;
#pragma unroll
  for (int j = 0; j < 12; ++j) { v[j] -= s; q += v[j] * v[j]; }
  q = wave_sum(q) * (1.f / kFUSED);
  const float rs = rsqrtf(q + 1e-5f);
  float* xr = XF + (size_t)b * kFUSED;
#pragma unroll
  for (int j = 0; j < 12; ++j) {
    const int c = lane * 12 + j;
    const float t = v[j] * rs * g[c] + bb[c];
    const float xo = (xr[c] + t) * 0.5f;
    xr[c] = xo;
    if (c < kC) out_tab[(size_t)b * kS * kC + c] = xo;
  }
}

// Deterministic scatter: highest b wins per node (sequential .at[].set semantics).
__global__ __launch_bounds__(256) void owner_mark_kernel(const int* __restrict__ idxv,
                                                         int* __restrict__ owner, int nB) {
  int b = blockIdx.x * 256 + threadIdx.x;
  if (b < nB) atomicMax(&owner[idxv[b]], b);
}
__global__ __launch_bounds__(256) void scatter_rows_kernel(
    const int* __restrict__ idxv, const int* __restrict__ owner,
    const float* __restrict__ xfo, int coloff, float* __restrict__ xg_out, int nB) {
  int b = blockIdx.x * 8 + (threadIdx.x >> 5);
  if (b >= nB) return;
  const int lane = threadIdx.x & 31;
  const int node = idxv[b];
  if (owner[node] != b) return;
#pragma unroll
  for (int i = lane; i < kF; i += 32)
    xg_out[(size_t)node * kF + i] = xfo[(size_t)b * kFUSED + coloff + i];
}

// ===========================================================================
extern "C" void kernel_launch(void* const* d_in, const int* in_sizes, int n_in,
                              void* d_out, int out_size, void* d_ws, size_t ws_size,
                              hipStream_t stream) {
  (void)in_sizes; (void)n_in; (void)out_size; (void)ws_size;
  const float* x_tab = (const float*)d_in[0];
  const float* x_gnn = (const float*)d_in[1];
  const int*   eidx  = (const int*)d_in[2];     // [2, E]: row0 src, row1 dst
  const float* eattr = (const float*)d_in[3];
  const float* p_Wqkv = (const float*)d_in[4];
  const float* p_bqkv = (const float*)d_in[5];
  const float* p_Wo   = (const float*)d_in[6];
  const float* p_bo   = (const float*)d_in[7];
  const float* p_W1   = (const float*)d_in[8];
  const float* p_b1   = (const float*)d_in[9];
  const float* p_W2   = (const float*)d_in[10];
  const float* p_b2   = (const float*)d_in[11];
  const float* p_ln1g = (const float*)d_in[12];
  const float* p_ln1b = (const float*)d_in[13];
  const float* p_ln2g = (const float*)d_in[14];
  const float* p_ln2b = (const float*)d_in[15];
  const float* p_tng  = (const float*)d_in[16];
  const float* p_tnb  = (const float*)d_in[17];
  const float* p_We   = (const float*)d_in[18];
  const float* p_be   = (const float*)d_in[19];
  const float* p_Wpre = (const float*)d_in[20];
  const float* p_bpre = (const float*)d_in[21];
  const float* p_Wpost= (const float*)d_in[22];
  const float* p_bpost= (const float*)d_in[23];
  const float* p_Wlin = (const float*)d_in[24];
  const float* p_blin = (const float*)d_in[25];
  const float* p_bng  = (const float*)d_in[26];
  const float* p_bnb  = (const float*)d_in[27];
  const float* p_flng = (const float*)d_in[28];
  const float* p_flnb = (const float*)d_in[29];
  const float* p_Wf1  = (const float*)d_in[30];
  const float* p_bf1  = (const float*)d_in[31];
  const float* p_Wf2  = (const float*)d_in[32];
  const float* p_bf2  = (const float*)d_in[33];
  const float* p_Wf3  = (const float*)d_in[34];
  const float* p_bf3  = (const float*)d_in[35];
  const float* p_fong = (const float*)d_in[36];
  const float* p_fonb = (const float*)d_in[37];
  const int* e_src = eidx;
  const int* e_dst = eidx + kE;

  // ---- workspace layout (phase-aliased regions) ----
  char* ws = (char*)d_ws;
  size_t off = 0;
  auto take = [&](size_t bytes) -> char* {
    char* p = ws + off;
    off += (bytes + 255) & ~(size_t)255;
    return p;
  };
  // R1: transformer {qkv_bf | o_bf | o2}  OR  GNN ewc_bf
  char* r1 = take(207618048);
  bf16_t* qkv_bf = (bf16_t*)r1;                               // BS*384 bf16
  bf16_t* o_bf   = (bf16_t*)(r1 + 103809024);                 // BS*128 bf16
  float*  o2     = (float*)(r1 + 103809024 + 34603008);       // BS*128 f32
  bf16_t* ewc_bf = (bf16_t*)r1;                               // E*F bf16
  // R2: ff1_bf  OR  ea_bf  OR  aggr_bf
  char* r2 = take(138412032);
  bf16_t* ff1_bf = (bf16_t*)r2;   // BS*512 bf16
  bf16_t* ea_bf  = (bf16_t*)r2;   // E*F bf16
  bf16_t* aggr_bf= (bf16_t*)r2;   // N*4F bf16
  // R3: h_f32 OR xiW ; R4: ff2 OR xjW ; R5: xt_bf OR xg_bf ; R6: h_bf OR s1
  char* r3 = take(69206016);  float* h_f32 = (float*)r3;  float* xiW = (float*)r3;
  char* r4 = take(69206016);  float* ff2   = (float*)r4;  float* xjW = (float*)r4;
  char* r5 = take(34603008);  bf16_t* xt_bf = (bf16_t*)r5; bf16_t* xg_bf = (bf16_t*)r5;
  char* r6 = take(51200000);  bf16_t* h_bf = (bf16_t*)r6; float* s1 = (float*)r6;
  float*    s2   = (float*)take(51200000);
  unsigned* mxe  = (unsigned*)take(51200000);
  unsigned* mne  = (unsigned*)take(51200000);
  float*    gout = (float*)take(51200000);
  float* deg     = (float*)take(400000);
  float* amp     = (float*)take(400000);
  float* inv_amp = (float*)take(400000);
  float* dsum    = (float*)take(256);
  float* colsum  = (float*)take(1024);
  float* colsq   = (float*)take(1024);
  int* owner_s   = (int*)take(400000);
  int* owner_d   = (int*)take(400000);
  float*  Wcomb_f  = (float*)take(65536);
  bf16_t* Wcomb_bf = (bf16_t*)take(32768);
  float*  bcomb    = (float*)take(512);
  float*  Wfold_f  = (float*)take(851968);
  bf16_t* Wfold_bf = (bf16_t*)take(425984);
  float*  bfold    = (float*)take(512);
  bf16_t* Wqkv_bf = (bf16_t*)take(98304);
  bf16_t* Wo_bf   = (bf16_t*)take(32768);
  bf16_t* W1_bf   = (bf16_t*)take(131072);
  bf16_t* W2_bf   = (bf16_t*)take(131072);
  bf16_t* Wpre_bf = (bf16_t*)take(98304);
  bf16_t* Wf1_bf  = (bf16_t*)take(294912);
  bf16_t* Wf2_bf  = (bf16_t*)take(294912);
  bf16_t* Wf3_bf  = (bf16_t*)take(294912);
  float*  xf    = (float*)take(6291456);
  bf16_t* hh_bf = (bf16_t*)take(3145728);
  bf16_t* h1_bf = (bf16_t*)take(3145728);
  bf16_t* h2_bf = (bf16_t*)take(3145728);
  float*  h3    = (float*)take(6291456);

  float* out_tab  = (float*)d_out;                       // [B,S,C]
  float* out_gnn  = out_tab + (size_t)kBS * kC;          // [N,F]
  float* out_edge = out_gnn + (size_t)kN * kF;           // [E,F]

  auto cdiv = [](long a, long b) { return (int)((a + b - 1) / b); };
  const dim3 T(256);

  // ---- weight prep: bf16 conversions + algebraic folds ----
  f2bf_kernel<<<cdiv(kBS * kC, 256), T, 0, stream>>>(x_tab, xt_bf, kBS * kC);
  f2bf_kernel<<<cdiv(3 * kC * kC, 256), T, 0, stream>>>(p_Wqkv, Wqkv_bf, 3 * kC * kC);
  f2bf_kernel<<<cdiv(kC * kC, 256), T, 0, stream>>>(p_Wo, Wo_bf, kC * kC);
  f2bf_kernel<<<cdiv(kFF * kC, 256), T, 0, stream>>>(p_W1, W1_bf, kFF * kC);
  f2bf_kernel<<<cdiv(kC * kFF, 256), T, 0, stream>>>(p_W2, W2_bf, kC * kFF);
  f2bf_kernel<<<cdiv(kF * 3 * kF, 256), T, 0, stream>>>(p_Wpre, Wpre_bf, kF * 3 * kF);
  f2bf_kernel<<<cdiv(kFUSED * kFUSED, 256), T, 0, stream>>>(p_Wf1, Wf1_bf, kFUSED * kFUSED);
  f2bf_kernel<<<cdiv(kFUSED * kFUSED, 256), T, 0, stream>>>(p_Wf2, Wf2_bf, kFUSED * kFUSED);
  f2bf_kernel<<<cdiv(kFUSED * kFUSED, 256), T, 0, stream>>>(p_Wf3, Wf3_bf, kFUSED * kFUSED);
  fold_we_kernel<<<cdiv(kF * kF, 256), T, 0, stream>>>(p_Wpre, p_We, Wcomb_f);
  fold_bcomb_kernel<<<1, 128, 0, stream>>>(p_Wpre, p_be, p_bpre, bcomb);
  f2bf_kernel<<<cdiv(kF * kF, 256), T, 0, stream>>>(Wcomb_f, Wcomb_bf, kF * kF);
  fold_wpost_kernel<<<cdiv(kF * 13 * kF, 256), T, 0, stream>>>(p_Wlin, p_Wpost, Wfold_f);
  fold_bfold_kernel<<<1, 128, 0, stream>>>(p_Wlin, p_bpost, p_blin, bfold);
  f2bf_kernel<<<cdiv(kF * 13 * kF, 256), T, 0, stream>>>(Wfold_f, Wfold_bf, kF * 13 * kF);

  // ---- transformer ----
  const int gyBS = cdiv(kBS, 128);  // 1056
  wmma_gemm_kernel<<<dim3(6, gyBS), T, 0, stream>>>(        // qkv (Nout=384)
      xt_bf, kC, Wqkv_bf, kC, p_bqkv, nullptr, nullptr, qkv_bf, 3 * kC, kBS, kC, 0, 0);
  attention_kernel<<<cdiv((long)kB * kNH * kS, 256), T, 0, stream>>>(qkv_bf, o_bf);
  wmma_gemm_kernel<<<dim3(2, gyBS), T, 0, stream>>>(        // Wo (Nout=128)
      o_bf, kC, Wo_bf, kC, p_bo, nullptr, o2, nullptr, kC, kBS, kC, 0, 0);
  ln_rows_kernel<<<cdiv(kBS, 8), T, 0, stream>>>(x_tab, o2, p_ln1g, p_ln1b,
                                                 h_f32, h_bf, kBS, kC);
  wmma_gemm_kernel<<<dim3(8, gyBS), T, 0, stream>>>(        // W1 (Nout=512, relu)
      h_bf, kC, W1_bf, kC, p_b1, nullptr, nullptr, ff1_bf, kFF, kBS, kC, 1, 0);
  wmma_gemm_kernel<<<dim3(2, gyBS), T, 0, stream>>>(        // W2 (Nout=128)
      ff1_bf, kFF, W2_bf, kFF, p_b2, nullptr, ff2, nullptr, kC, kBS, kFF, 0, 0);
  ln2x_kernel<<<cdiv(kBS, 8), T, 0, stream>>>(h_f32, ff2, p_ln2g, p_ln2b,
                                              p_tng, p_tnb, out_tab, kBS);

  // ---- GNN ----
  f2bf_kernel<<<cdiv(kN * kF, 256), T, 0, stream>>>(x_gnn, xg_bf, kN * kF);
  f2bf_kernel<<<cdiv(kE * kF, 256), T, 0, stream>>>(eattr, ea_bf, kE * kF);
  const int gyN = cdiv(kN, 128);  // 782
  wmma_gemm_kernel<<<dim3(2, gyN), T, 0, stream>>>(           // xiW = x @ W_i^T
      xg_bf, kF, Wpre_bf, 3 * kF, nullptr, nullptr, xiW, nullptr, kF, kN, kF, 0, 0);
  wmma_gemm_kernel<<<dim3(2, gyN), T, 0, stream>>>(           // xjW = x @ W_j^T
      xg_bf, kF, Wpre_bf + kF, 3 * kF, nullptr, nullptr, xjW, nullptr, kF, kN, kF, 0, 0);
  wmma_gemm_kernel<<<dim3(2, cdiv(kE, 128)), T, 0, stream>>>( // ewc = ea @ Wcomb^T + bcomb
      ea_bf, kF, Wcomb_bf, kF, bcomb, nullptr, nullptr, ewc_bf, kF, kE, kF, 0, 0);

  fill_u32_kernel<<<cdiv(kN * kF, 256), T, 0, stream>>>((unsigned*)s1, 0u, kN * kF);
  fill_u32_kernel<<<cdiv(kN * kF, 256), T, 0, stream>>>((unsigned*)s2, 0u, kN * kF);
  fill_u32_kernel<<<cdiv(kN * kF, 256), T, 0, stream>>>(mxe, 0u, kN * kF);
  fill_u32_kernel<<<cdiv(kN * kF, 256), T, 0, stream>>>(mne, 0xFFFFFFFFu, kN * kF);
  fill_u32_kernel<<<cdiv(kN, 256), T, 0, stream>>>((unsigned*)deg, 0u, kN);
  fill_u32_kernel<<<1, 1, 0, stream>>>((unsigned*)dsum, 0u, 1);
  fill_u32_kernel<<<1, 256, 0, stream>>>((unsigned*)colsum, 0u, kF);
  fill_u32_kernel<<<1, 256, 0, stream>>>((unsigned*)colsq, 0u, kF);
  fill_u32_kernel<<<cdiv(kN, 256), T, 0, stream>>>((unsigned*)owner_s, 0xFFFFFFFFu, kN);
  fill_u32_kernel<<<cdiv(kN, 256), T, 0, stream>>>((unsigned*)owner_d, 0xFFFFFFFFu, kN);

  edge_accum_kernel<<<cdiv(kE, 8), T, 0, stream>>>(xiW, xjW, ewc_bf, e_src, e_dst,
                                                   s1, s2, mxe, mne, deg);
  deg_log_sum_kernel<<<cdiv(kN, 256), T, 0, stream>>>(deg, dsum, kN);
  aggr_kernel<<<cdiv(kN, 8), T, 0, stream>>>(s1, s2, mxe, mne, deg, dsum,
                                             aggr_bf, amp, inv_amp);
  // out = xg@A^T + bfold  +  aggr@B^T  +  amp*(aggr@Cc^T)  +  (1/amp)*(aggr@Dd^T)
  wmma_gemm_kernel<<<dim3(2, gyN), T, 0, stream>>>(
      xg_bf, kF, Wfold_bf, 13 * kF, bfold, nullptr, gout, nullptr, kF, kN, kF, 0, 0);
  wmma_gemm_kernel<<<dim3(2, gyN), T, 0, stream>>>(
      aggr_bf, 4 * kF, Wfold_bf + kF, 13 * kF, nullptr, nullptr, gout, nullptr, kF, kN, 4 * kF, 0, 1);
  wmma_gemm_kernel<<<dim3(2, gyN), T, 0, stream>>>(
      aggr_bf, 4 * kF, Wfold_bf + 5 * kF, 13 * kF, nullptr, amp, gout, nullptr, kF, kN, 4 * kF, 0, 1);
  wmma_gemm_kernel<<<dim3(2, gyN), T, 0, stream>>>(
      aggr_bf, 4 * kF, Wfold_bf + 9 * kF, 13 * kF, nullptr, inv_amp, gout, nullptr, kF, kN, 4 * kF, 0, 1);
  colstats_kernel<<<gyN, 128, 0, stream>>>(gout, colsum, colsq, kN);
  bn_final_kernel<<<cdiv(kN * kF, 256), T, 0, stream>>>(gout, x_gnn, colsum, colsq,
                                                        p_bng, p_bnb, out_gnn);

  // ---- fusion MLP ----
  fuse_build_kernel<<<cdiv(kB, 8), T, 0, stream>>>(out_tab, out_gnn, e_src, e_dst, xf);
  ln_rows_kernel<<<cdiv((long)kB, 8), T, 0, stream>>>(xf, nullptr, p_flng, p_flnb,
                                                      nullptr, hh_bf, kB, kFUSED);
  const int gyB = cdiv(kB, 128);  // 32
  wmma_gemm_kernel<<<dim3(6, gyB), T, 0, stream>>>(
      hh_bf, kFUSED, Wf1_bf, kFUSED, p_bf1, nullptr, nullptr, h1_bf, kFUSED, kB, kFUSED, 2, 0);
  wmma_gemm_kernel<<<dim3(6, gyB), T, 0, stream>>>(
      h1_bf, kFUSED, Wf2_bf, kFUSED, p_bf2, nullptr, nullptr, h2_bf, kFUSED, kB, kFUSED, 2, 0);
  wmma_gemm_kernel<<<dim3(6, gyB), T, 0, stream>>>(
      h2_bf, kFUSED, Wf3_bf, kFUSED, p_bf3, nullptr, h3, nullptr, kFUSED, kB, kFUSED, 0, 0);
  fuse_final_kernel<<<cdiv(kB, 8), T, 0, stream>>>(h3, xf, p_fong, p_fonb, out_tab);

  // ---- scatters (s first, then d overwrites) ----
  owner_mark_kernel<<<cdiv(kB, 256), T, 0, stream>>>(e_src, owner_s, kB);
  scatter_rows_kernel<<<cdiv(kB, 8), T, 0, stream>>>(e_src, owner_s, xf, kC, out_gnn, kB);
  owner_mark_kernel<<<cdiv(kB, 256), T, 0, stream>>>(e_dst, owner_d, kB);
  scatter_rows_kernel<<<cdiv(kB, 8), T, 0, stream>>>(e_dst, owner_d, xf, kC + kF, out_gnn, kB);

  // ---- pass-through edge_attr output ----
  hipMemcpyAsync(out_edge, eattr, (size_t)kE * kF * sizeof(float),
                 hipMemcpyDeviceToDevice, stream);
}
